// RoseTea_21406117003313
// MI455X (gfx1250) — matrix operand order repaired and checked by reading, not compiled, not run on hardware
//
#include <hip/hip_runtime.h>
#include <hip/hip_bf16.h>

#define DD      256
#define NHEAD   8
#define DH      32
#define LMAXQ   64
#define BB      32
#define TTOT    1024
#define NTILES  16384
#define DIMG    512
#define NLEAVES 4096
#define TILEK   10
#define PPT     32
#define HTOT    2048
#define STOT    4096
#define NEDGE   16384
#define SMAX    128
#define HMAX    64
#define CANDN   320
#define ARC_S   30.0f

typedef __bf16 bf16_t;
typedef __attribute__((ext_vector_type(16))) __bf16 v16bf;
typedef __attribute__((ext_vector_type(8)))  float  v8f;

// ---------------------------------------------------------------- helpers

__device__ __forceinline__ bf16_t f2bf(float f) {
  union { float f; unsigned u; } x; x.f = f;
  unsigned r = x.u + 0x7FFFu + ((x.u >> 16) & 1u);   // round-to-nearest-even
  unsigned short hi = (unsigned short)(r >> 16);
  bf16_t o; __builtin_memcpy(&o, &hi, sizeof(o));
  return o;
}

// A fragment: 16x32 bf16, lane = row, K = e + 16h
__device__ __forceinline__ v16bf frag_a(const bf16_t* base, int ld) {
  int lane = threadIdx.x & 31;
  int h = lane >> 4, l = lane & 15;
  const bf16_t* row = base + l * ld;
  v16bf v;
#pragma unroll
  for (int e = 0; e < 16; ++e) {
    int k = e + 16 * h;
    v[e] = row[k];
  }
  return v;
}

// B fragment from k-major tile [32][ld]: lane = column, K = e + 16h
__device__ __forceinline__ v16bf frag_b_kmajor(const bf16_t* base, int col0, int ld) {
  int lane = threadIdx.x & 31;
  int h = lane >> 4, c = col0 + (lane & 15);
  v16bf v;
#pragma unroll
  for (int e = 0; e < 16; ++e) v[e] = base[(e + 16 * h) * ld + c];
  return v;
}

// B fragment from n-major tile [N][ld] (i.e. B^T rows contiguous in K)
__device__ __forceinline__ v16bf frag_b_nmajor(const bf16_t* base, int ld) {
  int lane = threadIdx.x & 31;
  int h = lane >> 4, l = lane & 15;
  const bf16_t* row = base + l * ld;
  v16bf v;
#pragma unroll
  for (int e = 0; e < 16; ++e) v[e] = row[e + 16 * h];
  return v;
}

__device__ __forceinline__ v8f wmma_bf16(v16bf a, v16bf b, v8f c) {
  return __builtin_amdgcn_wmma_f32_16x16x32_bf16(false, a, false, b, (short)0, c,
                                                 false, false);
}

// ---------------------------------------------------------------- GEMM NN
// C[M,N] = act(A[M,K] @ B[K,N]) ; block tile 64x128, wave tile 32x64.
// ACT: 0 = identity, 1 = tanh, 2 = relu.
template <int ACT>
__global__ __launch_bounds__(128) void gemm_nn_kernel(
    const float* __restrict__ A, const float* __restrict__ Bm,
    float* __restrict__ C, int M, int N, int K) {
  __shared__ bf16_t As[64 * 32];
  __shared__ bf16_t Bs[32 * 128];
  int tid = threadIdx.x;
  int m0 = blockIdx.y * 64, n0 = blockIdx.x * 128;
  int wid = tid >> 5, lane = tid & 31;
  int wm = wid >> 1, wn = wid & 1;
  int h = lane >> 4, l = lane & 15;
  v8f zero = {};
  v8f acc[2][4];
#pragma unroll
  for (int i = 0; i < 2; ++i)
#pragma unroll
    for (int j = 0; j < 4; ++j) acc[i][j] = zero;

  for (int k0 = 0; k0 < K; k0 += 32) {
#pragma unroll
    for (int it = 0; it < 16; ++it) {
      int e = tid + it * 128;
      As[e] = f2bf(A[(size_t)(m0 + (e >> 5)) * K + k0 + (e & 31)]);
    }
#pragma unroll
    for (int it = 0; it < 32; ++it) {
      int e = tid + it * 128;
      Bs[e] = f2bf(Bm[(size_t)(k0 + (e >> 7)) * N + n0 + (e & 127)]);
    }
    __syncthreads();
    v16bf af[2], bf[4];
#pragma unroll
    for (int i = 0; i < 2; ++i) af[i] = frag_a(As + (32 * wm + 16 * i) * 32, 32);
#pragma unroll
    for (int j = 0; j < 4; ++j) bf[j] = frag_b_kmajor(Bs, 64 * wn + 16 * j, 128);
#pragma unroll
    for (int i = 0; i < 2; ++i)
#pragma unroll
      for (int j = 0; j < 4; ++j) acc[i][j] = wmma_bf16(af[i], bf[j], acc[i][j]);
    __syncthreads();
  }
#pragma unroll
  for (int i = 0; i < 2; ++i)
#pragma unroll
    for (int j = 0; j < 4; ++j)
#pragma unroll
      for (int r = 0; r < 8; ++r) {
        int m = m0 + 32 * wm + 16 * i + r + 8 * h;
        int n = n0 + 64 * wn + 16 * j + l;
        float v = acc[i][j][r];
        if (ACT == 1) v = tanhf(v);
        else if (ACT == 2) v = fmaxf(v, 0.f);
        C[(size_t)m * N + n] = v;
      }
}

// ---------------------------------------------------------------- GEMM NT + cosine
// C[M,N] = (A[M,K] @ B[N,K]^T) / max(rnorm[m]*cnorm[n], 1e-6)
__global__ __launch_bounds__(128) void gemm_nt_cos_kernel(
    const float* __restrict__ A, const float* __restrict__ Bm,
    float* __restrict__ C, int M, int N, int K,
    const float* __restrict__ rnorm, const float* __restrict__ cnorm) {
  __shared__ bf16_t As[64 * 32];
  __shared__ bf16_t Bs[128 * 32];
  int tid = threadIdx.x;
  int m0 = blockIdx.y * 64, n0 = blockIdx.x * 128;
  int wid = tid >> 5, lane = tid & 31;
  int wm = wid >> 1, wn = wid & 1;
  int h = lane >> 4, l = lane & 15;
  v8f zero = {};
  v8f acc[2][4];
#pragma unroll
  for (int i = 0; i < 2; ++i)
#pragma unroll
    for (int j = 0; j < 4; ++j) acc[i][j] = zero;

  for (int k0 = 0; k0 < K; k0 += 32) {
#pragma unroll
    for (int it = 0; it < 16; ++it) {
      int e = tid + it * 128;
      As[e] = f2bf(A[(size_t)(m0 + (e >> 5)) * K + k0 + (e & 31)]);
    }
#pragma unroll
    for (int it = 0; it < 32; ++it) {
      int e = tid + it * 128;
      Bs[e] = f2bf(Bm[(size_t)(n0 + (e >> 5)) * K + k0 + (e & 31)]);
    }
    __syncthreads();
    v16bf af[2], bf[4];
#pragma unroll
    for (int i = 0; i < 2; ++i) af[i] = frag_a(As + (32 * wm + 16 * i) * 32, 32);
#pragma unroll
    for (int j = 0; j < 4; ++j) bf[j] = frag_b_nmajor(Bs + (64 * wn + 16 * j) * 32, 32);
#pragma unroll
    for (int i = 0; i < 2; ++i)
#pragma unroll
      for (int j = 0; j < 4; ++j) acc[i][j] = wmma_bf16(af[i], bf[j], acc[i][j]);
    __syncthreads();
  }
#pragma unroll
  for (int i = 0; i < 2; ++i)
#pragma unroll
    for (int j = 0; j < 4; ++j)
#pragma unroll
      for (int r = 0; r < 8; ++r) {
        int m = m0 + 32 * wm + 16 * i + r + 8 * h;
        int n = n0 + 64 * wn + 16 * j + l;
        float den = fmaxf(rnorm[m] * cnorm[n], 1e-6f);
        C[(size_t)m * N + n] = acc[i][j][r] / den;
      }
}

// ---------------------------------------------------------------- attention
// One block per (batch, head). 4 waves; wave w owns 16 query rows.
// Scores kept in registers; softmax via shfl; P stored to LDS bf16 for P@V.
template <int LK>
__global__ __launch_bounds__(128) void attn_kernel(
    const float* __restrict__ Q, const float* __restrict__ Kx,
    const float* __restrict__ V, float* __restrict__ CTX,
    const int* __restrict__ klen, int causal) {
  __shared__ bf16_t qs[64 * 32];
  __shared__ bf16_t ks[LK * 32];
  __shared__ bf16_t vs[LK * 32];
  __shared__ bf16_t ab[64 * LK];
  int tid = threadIdx.x;
  int b = blockIdx.x / NHEAD, hd = blockIdx.x % NHEAD;

  for (int e = tid; e < 64 * 32; e += 128)
    qs[e] = f2bf(Q[(size_t)(b * 64 + (e >> 5)) * DD + hd * 32 + (e & 31)]);
  for (int e = tid; e < LK * 32; e += 128) {
    ks[e] = f2bf(Kx[(size_t)(b * LK + (e >> 5)) * DD + hd * 32 + (e & 31)]);
    vs[e] = f2bf(V[(size_t)(b * LK + (e >> 5)) * DD + hd * 32 + (e & 31)]);
  }
  __syncthreads();

  int wid = tid >> 5, lane = tid & 31;
  int h = lane >> 4, l = lane & 15;
  int m0 = 16 * wid;
  const int NT = LK / 16;
  v8f zero = {};
  v8f sacc[NT];
  v16bf aq = frag_a(qs + m0 * 32, 32);
#pragma unroll
  for (int t = 0; t < NT; ++t) {
    v16bf bk = frag_b_nmajor(ks + t * 16 * 32, 32);
    sacc[t] = wmma_bf16(aq, bk, zero);
  }
  int kl = klen[b];
  const float scale = 0.17677669529663687f;  // 1/sqrt(32)
#pragma unroll
  for (int t = 0; t < NT; ++t)
#pragma unroll
    for (int r = 0; r < 8; ++r) {
      int m = m0 + r + 8 * h;
      int n = t * 16 + l;
      float s = sacc[t][r] * scale;
      if (n >= kl || (causal && n > m)) s = -1e9f;
      sacc[t][r] = s;
    }
  // softmax per row (each row lives on 16 lanes of one half-wave)
#pragma unroll
  for (int r = 0; r < 8; ++r) {
    float mx = -1e30f;
#pragma unroll
    for (int t = 0; t < NT; ++t) mx = fmaxf(mx, sacc[t][r]);
#pragma unroll
    for (int o = 1; o < 16; o <<= 1) mx = fmaxf(mx, __shfl_xor(mx, o, 32));
    float sm = 0.f;
#pragma unroll
    for (int t = 0; t < NT; ++t) {
      float e = __expf(sacc[t][r] - mx);
      sacc[t][r] = e;
      sm += e;
    }
#pragma unroll
    for (int o = 1; o < 16; o <<= 1) sm += __shfl_xor(sm, o, 32);
    float inv = 1.f / sm;
#pragma unroll
    for (int t = 0; t < NT; ++t) {
      int m = m0 + r + 8 * h;
      int n = t * 16 + l;
      ab[m * LK + n] = f2bf(sacc[t][r] * inv);
    }
  }
  __syncthreads();
  // O = P @ V  (dh = 32 -> two 16-col tiles)
#pragma unroll
  for (int j = 0; j < 2; ++j) {
    v8f occ = zero;
#pragma unroll
    for (int kc = 0; kc < LK; kc += 32) {
      v16bf ap = frag_a(ab + m0 * LK + kc, LK);
      v16bf bv = frag_b_kmajor(vs + kc * 32, 16 * j, 32);
      occ = wmma_bf16(ap, bv, occ);
    }
#pragma unroll
    for (int r = 0; r < 8; ++r) {
      int m = m0 + r + 8 * h;
      int n = 16 * j + l;
      CTX[(size_t)(b * 64 + m) * DD + hd * 32 + n] = occ[r];
    }
  }
}

// ---------------------------------------------------------------- glue kernels

__global__ void zero_kernel(float* p, int n) {
  int i = blockIdx.x * 256 + threadIdx.x;
  if (i < n) p[i] = 0.f;
}

__global__ void setup_offsets_kernel(const int* tl, const int* hl, const int* sl,
                                     int* offT, int* offH, int* offS,
                                     int* r2b, int* r2t) {
  if (blockIdx.x == 0 && threadIdx.x == 0) {
    int a = 0, b = 0, c = 0;
    for (int i = 0; i < BB; ++i) {
      offT[i] = a; a += tl[i];
      offH[i] = b; b += hl[i];
      offS[i] = c; c += sl[i];
    }
    int r = 0;
    for (int i = 0; i < BB; ++i)
      for (int t = 0; t < tl[i] && r < TTOT; ++t) { r2b[r] = i; r2t[r] = t; ++r; }
  }
}

__global__ void gather_rows_kernel(const float* src, const int* idx, float* dst) {
  int i = blockIdx.x, d = threadIdx.x;
  dst[(size_t)i * DD + d] = src[(size_t)idx[i] * DD + d];
}

__global__ void edge_agg_kernel(const float* TNE, const int* nodes,
                                const int* edges, float* agg) {
  int e = blockIdx.x, d = threadIdx.x;
  int src = edges[e], dst = edges[NEDGE + e];
  atomicAdd(&agg[(size_t)dst * DD + d], TNE[(size_t)nodes[src] * DD + d]);
}

__global__ void nx_add_kernel(const float* TNE, const int* nodes, float* agg) {
  int s = blockIdx.x, d = threadIdx.x;
  agg[(size_t)s * DD + d] += TNE[(size_t)nodes[s] * DD + d];
}

__global__ void poihist_kernel(const float* xy, const float* Wpe, const float* bpe,
                               float* out) {
  int i = blockIdx.x, d = threadIdx.x;
  out[(size_t)i * DD + d] = xy[i * 2] * Wpe[d] + xy[i * 2 + 1] * Wpe[DD + d] + bpe[d];
}

__global__ void recover_pad_kernel(const float* src, const int* len, const int* off,
                                   float* dst, int Lm) {
  int blk = blockIdx.x;
  int b = blk / Lm, t = blk % Lm, d = threadIdx.x;
  float v = 0.f;
  if (t < len[b]) v = src[(size_t)(off[b] + t) * DD + d];
  dst[(size_t)blk * DD + d] = v;
}

__global__ void build_te_kernel(const float* TNE, const int* leaves_seq,
                                const int* pos_seq, const float* poi_traj,
                                const float* tstamp, const float* pos_table,
                                const float* wt1, const float* wt2,
                                const float* Wpe, const float* bpe,
                                const int* tlen, const int* toff,
                                float* TE, float* PEt) {
  int blk = blockIdx.x;
  int b = blk >> 6, t = blk & 63, d = threadIdx.x;
  float lse = 0.f, pe = 0.f;
  if (t < tlen[b]) {
    int fi = toff[b] + t;
    lse = TNE[(size_t)leaves_seq[fi] * DD + d];
    pe = poi_traj[fi * 2] * Wpe[d] + poi_traj[fi * 2 + 1] * Wpe[DD + d] + bpe[d];
  }
  float ts = tstamp[blk];
  TE[(size_t)blk * DD + d]  = lse + pos_table[(size_t)pos_seq[blk] * DD + d] + ts * wt1[d];
  PEt[(size_t)blk * DD + d] = pe + ts * wt2[d];
}

__global__ __launch_bounds__(256) void add_ln_kernel(const float* X, const float* Y,
                                                     float* O, int rows) {
  int wid = threadIdx.x >> 5, lane = threadIdx.x & 31;
  int row = blockIdx.x * 8 + wid;
  if (row >= rows) return;
  const float* x = X + (size_t)row * DD;
  const float* y = Y + (size_t)row * DD;
  float v[8], s = 0.f, s2 = 0.f;
#pragma unroll
  for (int j = 0; j < 8; ++j) {
    v[j] = x[lane + 32 * j] + y[lane + 32 * j];
    s += v[j]; s2 += v[j] * v[j];
  }
  for (int o = 1; o < 32; o <<= 1) { s += __shfl_xor(s, o, 32); s2 += __shfl_xor(s2, o, 32); }
  float mean = s * (1.f / DD);
  float var = s2 * (1.f / DD) - mean * mean;
  float inv = rsqrtf(var + 1e-5f);
  float* out = O + (size_t)row * DD;
#pragma unroll
  for (int j = 0; j < 8; ++j) out[lane + 32 * j] = (v[j] - mean) * inv;
}

__global__ __launch_bounds__(256) void rownorm_kernel(const float* X, float* nrm, int rows) {
  int wid = threadIdx.x >> 5, lane = threadIdx.x & 31;
  int row = blockIdx.x * 8 + wid;
  if (row >= rows) return;
  const float* x = X + (size_t)row * DD;
  float s = 0.f;
#pragma unroll
  for (int j = 0; j < 8; ++j) { float a = x[lane + 32 * j]; s += a * a; }
  for (int o = 1; o < 32; o <<= 1) s += __shfl_xor(s, o, 32);
  if (lane == 0) nrm[row] = sqrtf(s);
}

__global__ void pick_kernel(const float* src, const int* r2b, const int* r2t, float* dst) {
  int i = blockIdx.x, d = threadIdx.x;
  dst[(size_t)i * DD + d] = src[(size_t)(r2b[i] * 64 + r2t[i]) * DD + d];
}

__global__ void topk_kernel(const float* prob, const int* target, int* idx) {
  int n = blockIdx.x * blockDim.x + threadIdx.x;
  if (n >= TTOT) return;
  float tv[TILEK]; int ti[TILEK];
#pragma unroll
  for (int k = 0; k < TILEK; ++k) { tv[k] = -1e30f; ti[k] = 0; }
  const float* row = prob + (size_t)n * NLEAVES;
  for (int j = 0; j < NLEAVES; ++j) {
    float v = row[j];
    if (v > tv[TILEK - 1]) {
      int p = TILEK - 1;
      while (p > 0 && tv[p - 1] < v) { tv[p] = tv[p - 1]; ti[p] = ti[p - 1]; --p; }
      tv[p] = v; ti[p] = j;
    }
  }
  int tgt = target[n];
  bool hit = false;
#pragma unroll
  for (int k = 0; k < TILEK; ++k) hit = hit || (ti[k] == tgt);
  if (!hit) ti[TILEK - 1] = tgt;
#pragma unroll
  for (int k = 0; k < TILEK; ++k) idx[n * TILEK + k] = ti[k];
}

__global__ void cand_kernel(const int* idx, const int* tpt, const int* tpoi,
                            int* candc, int* cnt_out, int* tpi_out, float* out_tpi) {
  int n = blockIdx.x * blockDim.x + threadIdx.x;
  if (n >= TTOT) return;
  int tp0 = tpoi[n * 2];
  int position = 0; bool found = false;
  for (int j = 0; j < CANDN; ++j) {
    int ti = idx[n * TILEK + (j >> 5)];
    int s = j & 31;
    int p0 = tpt[((size_t)ti * PPT + s) * 2];
    if (!found && p0 == tp0) { position = j; found = true; }
  }
  int cnt = 0, before = 0;
  for (int j = 0; j < CANDN; ++j) {
    int ti = idx[n * TILEK + (j >> 5)];
    int s = j & 31;
    int p0 = tpt[((size_t)ti * PPT + s) * 2];
    int p1 = tpt[((size_t)ti * PPT + s) * 2 + 1];
    bool valid = (p0 != -1) && (p1 != -1);
    if (valid) {
      candc[((size_t)n * CANDN + cnt) * 2] = p0;
      candc[((size_t)n * CANDN + cnt) * 2 + 1] = p1;
      if (j < position) ++before;
      ++cnt;
    }
  }
  for (int c = cnt; c < CANDN; ++c) {
    candc[((size_t)n * CANDN + c) * 2] = 0;
    candc[((size_t)n * CANDN + c) * 2 + 1] = 0;
  }
  cnt_out[n] = cnt;
  tpi_out[n] = before;
  out_tpi[n] = (float)before;
}

__global__ __launch_bounds__(32) void poicos_kernel(const int* candc, const int* cnt_in,
                                                    const float* Wpe, const float* bpe,
                                                    const float* pout, const float* pnrm,
                                                    float* pcos, float* out_prob) {
  int n = blockIdx.x;
  int c = blockIdx.y * 32 + threadIdx.x;
  int cnt = cnt_in[n];
  float num = 0.f, nr = 0.f;
  if (c < cnt) {
    float p0 = (float)candc[((size_t)n * CANDN + c) * 2];
    float p1 = (float)candc[((size_t)n * CANDN + c) * 2 + 1];
    const float* po = pout + (size_t)n * DD;
    for (int d = 0; d < DD; ++d) {
      float cp = p0 * Wpe[d] + p1 * Wpe[DD + d] + bpe[d];
      num += cp * po[d];
      nr  += cp * cp;
    }
  }
  float den = fmaxf(sqrtf(nr) * pnrm[n], 1e-6f);
  float cs = num / den;
  pcos[(size_t)n * CANDN + c] = cs;
  out_prob[(size_t)n * CANDN + c] = (c < cnt) ? cs : -1.f;
}

__global__ __launch_bounds__(256) void arc_tile_kernel(const float* prob,
                                                       const int* target, float* part) {
  int wid = threadIdx.x >> 5, lane = threadIdx.x & 31;
  int row = blockIdx.x * 8 + wid;
  if (row >= TTOT) return;
  const float* p = prob + (size_t)row * NLEAVES;
  int tgt = target[row];
  float t = fminf(fmaxf(p[tgt], -1.f + 1e-7f), 1.f - 1e-7f);
  float tm = cosf(acosf(t) + 0.05f);
  float lt = ARC_S * tm;
  float mx = -1e30f;
  for (int j = lane; j < NLEAVES; j += 32) {
    float lj = (j == tgt) ? lt : ARC_S * p[j];
    mx = fmaxf(mx, lj);
  }
  for (int o = 1; o < 32; o <<= 1) mx = fmaxf(mx, __shfl_xor(mx, o, 32));
  float sm = 0.f;
  for (int j = lane; j < NLEAVES; j += 32) {
    float lj = (j == tgt) ? lt : ARC_S * p[j];
    sm += __expf(lj - mx);
  }
  for (int o = 1; o < 32; o <<= 1) sm += __shfl_xor(sm, o, 32);
  if (lane == 0) part[row] = mx + logf(sm) - lt;
}

__global__ __launch_bounds__(256) void arc_poi_kernel(const float* pcos, const int* cnt_in,
                                                      const int* tpi, float* part) {
  int wid = threadIdx.x >> 5, lane = threadIdx.x & 31;
  int row = blockIdx.x * 8 + wid;
  if (row >= TTOT) return;
  const float* p = pcos + (size_t)row * CANDN;
  int tgt = tpi[row];
  int cnt = cnt_in[row];
  float t = fminf(fmaxf(p[tgt], -1.f + 1e-7f), 1.f - 1e-7f);
  float tm = cosf(acosf(t) + 0.1f);
  float lt = ARC_S * tm;
  float mx = -1e30f;
  for (int j = lane; j < CANDN; j += 32) {
    float lj = (j >= cnt) ? -1e9f : ((j == tgt) ? lt : ARC_S * p[j]);
    mx = fmaxf(mx, lj);
  }
  for (int o = 1; o < 32; o <<= 1) mx = fmaxf(mx, __shfl_xor(mx, o, 32));
  float sm = 0.f;
  for (int j = lane; j < CANDN; j += 32) {
    float lj = (j >= cnt) ? -1e9f : ((j == tgt) ? lt : ARC_S * p[j]);
    sm += __expf(lj - mx);
  }
  for (int o = 1; o < 32; o <<= 1) sm += __shfl_xor(sm, o, 32);
  if (lane == 0) part[row] = mx + logf(sm) - lt;
}

__global__ __launch_bounds__(256) void mean_kernel(const float* part, float* out) {
  __shared__ float sh[256];
  float s = 0.f;
  for (int i = threadIdx.x; i < TTOT; i += 256) s += part[i];
  sh[threadIdx.x] = s;
  __syncthreads();
  for (int o = 128; o > 0; o >>= 1) {
    if (threadIdx.x < o) sh[threadIdx.x] += sh[threadIdx.x + o];
    __syncthreads();
  }
  if (threadIdx.x == 0) out[0] = sh[0] / (float)TTOT;
}

// ---------------------------------------------------------------- launch

extern "C" void kernel_launch(void* const* d_in, const int* in_sizes, int n_in,
                              void* d_out, int out_size, void* d_ws, size_t ws_size,
                              hipStream_t stream) {
  (void)in_sizes; (void)n_in; (void)out_size; (void)ws_size;
  const int*   traj_len   = (const int*)d_in[0];
  const int*   leaves_seq = (const int*)d_in[1];
  const int*   pos_seq    = (const int*)d_in[2];
  const int*   hist_len   = (const int*)d_in[3];
  const int*   sub_len    = (const int*)d_in[4];
  const int*   sg_nodes   = (const int*)d_in[5];
  const int*   sg_edges   = (const int*)d_in[6];
  const int*   tgt_tile   = (const int*)d_in[7];
  const int*   tgt_poi    = (const int*)d_in[8];
  const int*   tpt        = (const int*)d_in[9];
  const int*   leaves_id  = (const int*)d_in[10];
  const float* imagery    = (const float*)d_in[11];
  const float* poi_traj   = (const float*)d_in[12];
  const float* hist_traj  = (const float*)d_in[13];
  const float* tstamp     = (const float*)d_in[14];
  const float* pos_table  = (const float*)d_in[15];
  const float* w_time     = (const float*)d_in[16];
  const float* w_time2    = (const float*)d_in[17];
  const float* W_ee       = (const float*)d_in[18];
  const float* W_pe       = (const float*)d_in[19];
  const float* b_pe       = (const float*)d_in[20];
  const float* W_qrp_t    = (const float*)d_in[21];
  const float* W_qrp_p    = (const float*)d_in[22];
  const float* Wq_s       = (const float*)d_in[23];
  const float* Wk_s       = (const float*)d_in[24];
  const float* Wv_s       = (const float*)d_in[25];
  const float* Wo_s       = (const float*)d_in[26];
  const float* Wq_c       = (const float*)d_in[27];
  const float* Wk_c       = (const float*)d_in[28];
  const float* Wv_c       = (const float*)d_in[29];
  const float* Wo_c       = (const float*)d_in[30];
  const float* W1         = (const float*)d_in[31];
  const float* W2         = (const float*)d_in[32];

  // ---- workspace layout
  float* f = (float*)d_ws;
  size_t o = 0;
  auto alloc = [&](size_t n) { float* p = f + o; o += n; return p; };
  float* TNE  = alloc((size_t)NTILES * DD);
  float* LEb  = alloc((size_t)NLEAVES * DD);
  float* AGG  = alloc((size_t)STOT * DD);      // becomes node_x + agg
  float* SUBT = alloc((size_t)STOT * DD);
  float* POIH = alloc((size_t)HTOT * DD);
  float* SUBP = alloc((size_t)HTOT * DD);
  float* STEb = alloc((size_t)BB * SMAX * DD);
  float* SPEb = alloc((size_t)BB * HMAX * DD);
  float* TE   = alloc((size_t)BB * LMAXQ * DD);
  float* PEt  = alloc((size_t)BB * LMAXQ * DD);
  float* Xb   = alloc((size_t)2048 * DD);
  float* Qb   = alloc((size_t)2048 * DD);
  float* Kb   = alloc((size_t)4096 * DD);
  float* Vb   = alloc((size_t)4096 * DD);
  float* CTX  = alloc((size_t)2048 * DD);
  float* MHO  = alloc((size_t)2048 * DD);
  float* H1   = alloc((size_t)2048 * 4 * DD);
  float* TOUT = alloc((size_t)TTOT * DD);
  float* POUT = alloc((size_t)TTOT * DD);
  float* NLE  = alloc(NLEAVES);
  float* NTO  = alloc(TTOT);
  float* NPO  = alloc(TTOT);
  float* PCOS = alloc((size_t)TTOT * CANDN);
  float* PART = alloc(TTOT);
  int* ip = (int*)(f + o);
  int* offT = ip;            ip += BB;
  int* offH = ip;            ip += BB;
  int* offS = ip;            ip += BB;
  int* r2b  = ip;            ip += TTOT;
  int* r2t  = ip;            ip += TTOT;
  int* IDX  = ip;            ip += TTOT * TILEK;
  int* CANDC = ip;           ip += TTOT * CANDN * 2;
  int* CNT  = ip;            ip += TTOT;
  int* TPI  = ip;            ip += TTOT;

  float* out_tile_prob = (float*)d_out;
  float* out_tile_loss = out_tile_prob + (size_t)TTOT * NLEAVES;
  float* out_poi_prob  = out_tile_loss + 1;
  float* out_poi_loss  = out_poi_prob + (size_t)TTOT * CANDN;
  float* out_tpi       = out_poi_loss + 1;

  auto gemm = [&](const float* A, const float* Bm, float* C, int M, int N, int K,
                  int act) {
    dim3 g(N / 128, M / 64);
    if (act == 1)      gemm_nn_kernel<1><<<g, 128, 0, stream>>>(A, Bm, C, M, N, K);
    else if (act == 2) gemm_nn_kernel<2><<<g, 128, 0, stream>>>(A, Bm, C, M, N, K);
    else               gemm_nn_kernel<0><<<g, 128, 0, stream>>>(A, Bm, C, M, N, K);
  };

  // ---- embeddings / graph
  setup_offsets_kernel<<<1, 1, 0, stream>>>(traj_len, hist_len, sub_len,
                                            offT, offH, offS, r2b, r2t);
  gemm(imagery, W_ee, TNE, NTILES, DD, DIMG, 0);
  gather_rows_kernel<<<NLEAVES, DD, 0, stream>>>(TNE, leaves_id, LEb);
  zero_kernel<<<(STOT * DD) / 256, 256, 0, stream>>>(AGG, STOT * DD);
  edge_agg_kernel<<<NEDGE, DD, 0, stream>>>(TNE, sg_nodes, sg_edges, AGG);
  nx_add_kernel<<<STOT, DD, 0, stream>>>(TNE, sg_nodes, AGG);
  gemm(AGG, W_qrp_t, SUBT, STOT, DD, DD, 1);
  poihist_kernel<<<HTOT, DD, 0, stream>>>(hist_traj, W_pe, b_pe, POIH);
  gemm(POIH, W_qrp_p, SUBP, HTOT, DD, DD, 1);
  recover_pad_kernel<<<BB * SMAX, DD, 0, stream>>>(SUBT, sub_len, offS, STEb, SMAX);
  recover_pad_kernel<<<BB * HMAX, DD, 0, stream>>>(SUBP, hist_len, offH, SPEb, HMAX);
  build_te_kernel<<<BB * LMAXQ, DD, 0, stream>>>(TNE, leaves_seq, pos_seq, poi_traj,
                                                 tstamp, pos_table, w_time, w_time2,
                                                 W_pe, b_pe, traj_len, offT, TE, PEt);

  auto run_decoder = [&](const float* xin, const float* mem, int Lm, const int* memlen) {
    // self-attention block
    gemm(xin, Wq_s, Qb, 2048, DD, DD, 0);
    gemm(xin, Wk_s, Kb, 2048, DD, DD, 0);
    gemm(xin, Wv_s, Vb, 2048, DD, DD, 0);
    attn_kernel<64><<<BB * NHEAD, 128, 0, stream>>>(Qb, Kb, Vb, CTX, traj_len, 1);
    gemm(CTX, Wo_s, MHO, 2048, DD, DD, 0);
    add_ln_kernel<<<2048 / 8, 256, 0, stream>>>(xin, MHO, Xb, 2048);
    // cross-attention block
    gemm(Xb, Wq_c, Qb, 2048, DD, DD, 0);
    gemm(mem, Wk_c, Kb, BB * Lm, DD, DD, 0);
    gemm(mem, Wv_c, Vb, BB * Lm, DD, DD, 0);
    if (Lm == 128)
      attn_kernel<128><<<BB * NHEAD, 128, 0, stream>>>(Qb, Kb, Vb, CTX, memlen, 0);
    else
      attn_kernel<64><<<BB * NHEAD, 128, 0, stream>>>(Qb, Kb, Vb, CTX, memlen, 0);
    gemm(CTX, Wo_c, MHO, 2048, DD, DD, 0);
    add_ln_kernel<<<2048 / 8, 256, 0, stream>>>(Xb, MHO, Xb, 2048);
    // FFN block
    gemm(Xb, W1, H1, 2048, 4 * DD, DD, 2);
    gemm(H1, W2, MHO, 2048, DD, 4 * DD, 0);
    add_ln_kernel<<<2048 / 8, 256, 0, stream>>>(Xb, MHO, Xb, 2048);
  };

  // ---- tile branch
  run_decoder(TE, STEb, SMAX, sub_len);
  pick_kernel<<<TTOT, DD, 0, stream>>>(Xb, r2b, r2t, TOUT);
  rownorm_kernel<<<TTOT / 8, 256, 0, stream>>>(TOUT, NTO, TTOT);
  rownorm_kernel<<<NLEAVES / 8, 256, 0, stream>>>(LEb, NLE, NLEAVES);
  gemm_nt_cos_kernel<<<dim3(NLEAVES / 128, TTOT / 64), 128, 0, stream>>>(
      TOUT, LEb, out_tile_prob, TTOT, NLEAVES, DD, NTO, NLE);
  topk_kernel<<<TTOT / 256, 256, 0, stream>>>(out_tile_prob, tgt_tile, IDX);
  cand_kernel<<<TTOT / 256, 256, 0, stream>>>(IDX, tpt, tgt_poi, CANDC, CNT, TPI, out_tpi);

  // ---- poi branch
  run_decoder(PEt, SPEb, HMAX, hist_len);
  pick_kernel<<<TTOT, DD, 0, stream>>>(Xb, r2b, r2t, POUT);
  rownorm_kernel<<<TTOT / 8, 256, 0, stream>>>(POUT, NPO, TTOT);
  poicos_kernel<<<dim3(TTOT, CANDN / 32), 32, 0, stream>>>(CANDC, CNT, W_pe, b_pe,
                                                           POUT, NPO, PCOS, out_poi_prob);

  // ---- losses
  arc_tile_kernel<<<TTOT / 8, 256, 0, stream>>>(out_tile_prob, tgt_tile, PART);
  mean_kernel<<<1, 256, 0, stream>>>(PART, out_tile_loss);
  arc_poi_kernel<<<TTOT / 8, 256, 0, stream>>>(PCOS, CNT, TPI, PART);
  mean_kernel<<<1, 256, 0, stream>>>(PART, out_poi_loss);
}
